// LocalAttention1D_47966194762202
// MI455X (gfx1250) — matrix operand order repaired and checked by reading, not compiled, not run on hardware
//
#include <hip/hip_runtime.h>

#define DEVINL __device__ __forceinline__

typedef __attribute__((ext_vector_type(16))) __bf16        v16bf;
typedef __attribute__((ext_vector_type(8)))  float         v8f;
typedef __attribute__((ext_vector_type(4)))  unsigned int  v4u;

union BF16Tile {
    v16bf          v;
    v4u            q[2];
    unsigned short u[16];
};

DEVINL unsigned short f32_to_bf16(float f) {
    unsigned int u = __float_as_uint(f);
    unsigned int r = u + 0x7FFFu + ((u >> 16) & 1u);   // round-to-nearest-even
    return (unsigned short)(r >> 16);
}

// A-operand (16x32 bf16): lane holds row m = lane&15; h = lane>>4 selects
// K runs {h*8..h*8+7} and {16+h*8..16+h*8+7}. rowk points at row start + kbase.
DEVINL v16bf load_a_bf16(const unsigned short* rowk, int h) {
    BF16Tile t;
    t.q[0] = *reinterpret_cast<const v4u*>(rowk + h * 8);
    t.q[1] = *reinterpret_cast<const v4u*>(rowk + 16 + h * 8);
    return t.v;
}

// B-operand (32x16 bf16): lane holds column n = lane&15; h = lane>>4 selects
// K runs {h*8..h*8+7} and {16+h*8..16+h*8+7}. colk points at the column's K-vector.
DEVINL v16bf load_b_bf16(const unsigned short* colk, int h) {
    BF16Tile t;
    t.q[0] = *reinterpret_cast<const v4u*>(colk + h * 8);
    t.q[1] = *reinterpret_cast<const v4u*>(colk + 16 + h * 8);
    return t.v;
}

DEVINL v8f wmma_bf16(v16bf a, v16bf b, v8f c) {
    return __builtin_amdgcn_wmma_f32_16x16x32_bf16(
        /*neg_a=*/false, a, /*neg_b=*/false, b,
        /*c_mod=*/(short)0, c, /*reuse_a=*/false, /*reuse_b=*/false);
}

// ---------------------------------------------------------------------------
// Problem constants
// ---------------------------------------------------------------------------
#define LSEQ 4096
#define CDIM 512
#define NHEAD 8
#define HDIM 64
#define HALFW 64
#define ATTN_SCALE 0.125f   // 64^-0.5

// ---------------------------------------------------------------------------
// Kernel 0: one-shot bf16 conversions.
//   wqkvT[n][k] = bf16(w_qkv[k][n])   n<1536, k<512   (transposed)
//   wprojT[n][k] = bf16(w_proj[k][n]) n<512,  k<512   (transposed)
//   xbf[i]       = bf16(x[i])                          (straight copy)
// ---------------------------------------------------------------------------
__global__ void prep_inputs_kernel(const float* __restrict__ wqkv,
                                   const float* __restrict__ wproj,
                                   const float* __restrict__ x,
                                   unsigned short* __restrict__ wqkvT,
                                   unsigned short* __restrict__ wprojT,
                                   unsigned short* __restrict__ xbf) {
    const int NQKV  = 3 * CDIM * CDIM;        // 786432
    const int NPROJ = CDIM * CDIM;            // 262144
    int idx = blockIdx.x * blockDim.x + threadIdx.x;
    if (idx < NQKV) {
        int n = idx / CDIM, k = idx % CDIM;
        wqkvT[idx] = f32_to_bf16(wqkv[(size_t)k * (3 * CDIM) + n]);
    } else if (idx < NQKV + NPROJ) {
        int i = idx - NQKV;
        int n = i / CDIM, k = i % CDIM;
        wprojT[i] = f32_to_bf16(wproj[(size_t)k * CDIM + n]);
    } else {
        int i = idx - NQKV - NPROJ;           // < 2*4096*512
        xbf[i] = f32_to_bf16(x[i]);
    }
}

// ---------------------------------------------------------------------------
// Generic 32x64-per-wave bf16 GEMM body: 2 row-tiles x 4 col-tiles, K-loop of
// 32.  Loads A(2) + B(4) up front each step, then 8 WMMAs; prefetches the
// next K-slice of the A rows.
// ---------------------------------------------------------------------------
struct GemmAcc {
    v8f a0[4];   // rows rbase..rbase+15
    v8f a1[4];   // rows rbase+16..rbase+31
};

DEVINL void gemm_32x64(const unsigned short* __restrict__ arow0,
                       const unsigned short* __restrict__ arow1,
                       const unsigned short* __restrict__ wT,
                       int nbase, int m, int h, GemmAcc& g) {
#pragma unroll
    for (int t = 0; t < 4; ++t) { g.a0[t] = v8f{0,0,0,0,0,0,0,0}; g.a1[t] = v8f{0,0,0,0,0,0,0,0}; }
#pragma unroll 4
    for (int kb = 0; kb < CDIM; kb += 32) {
        v16bf a0 = load_a_bf16(arow0 + kb, h);
        v16bf a1 = load_a_bf16(arow1 + kb, h);
        __builtin_prefetch(arow0 + kb + 32, 0, 3);   // global_prefetch_b8
        __builtin_prefetch(arow1 + kb + 32, 0, 3);
        v16bf b[4];
#pragma unroll
        for (int t = 0; t < 4; ++t)
            b[t] = load_b_bf16(wT + (size_t)(nbase + t * 16 + m) * CDIM + kb, h);
#pragma unroll
        for (int t = 0; t < 4; ++t) {
            g.a0[t] = wmma_bf16(a0, b[t], g.a0[t]);
            g.a1[t] = wmma_bf16(a1, b[t], g.a1[t]);
        }
    }
}

// ---------------------------------------------------------------------------
// Kernel 1: QKV projection GEMM.  One wave computes a 32x64 tile of
// y = x @ w_qkv (rows = 8192, cols = 1536); scatters bf16 results into
// Q rows, K rows and V^T rows per head.
// ---------------------------------------------------------------------------
__global__ __launch_bounds__(32) void qkv_gemm_kernel(
    const unsigned short* __restrict__ xbf,   // [B*L, 512] bf16
    const unsigned short* __restrict__ wT,    // [1536, 512] bf16
    unsigned short* __restrict__ Qb,          // [B*H, L, 64]
    unsigned short* __restrict__ Kb,          // [B*H, L, 64]
    unsigned short* __restrict__ Vt)          // [B*H, 64, L]
{
    const int lane = threadIdx.x;
    const int m = lane & 15, h = lane >> 4;
    const int rt    = blockIdx.x / 24;        // 24 = 1536/64 col strips
    const int nt    = blockIdx.x % 24;
    const int rbase = rt * 32;
    const int nbase = nt * 64;

    GemmAcc g;
    gemm_32x64(xbf + (size_t)(rbase + m) * CDIM,
               xbf + (size_t)(rbase + 16 + m) * CDIM,
               wT, nbase, m, h, g);

#pragma unroll
    for (int half = 0; half < 2; ++half) {
#pragma unroll
        for (int t = 0; t < 4; ++t) {
#pragma unroll
            for (int v = 0; v < 8; ++v) {
                float val = half ? g.a1[t][v] : g.a0[t][v];
                int r  = rbase + half * 16 + v + 8 * h;   // global row
                int bb = r >> 12;                          // / LSEQ
                int l  = r & (LSEQ - 1);
                int c3 = nbase + t * 16 + m;
                int s  = c3 >> 9;                          // 0:q 1:k 2:v
                int hd = c3 & 511;
                int hh = hd >> 6;
                int d  = hd & 63;
                unsigned short bv = f32_to_bf16(val);
                size_t headidx = (size_t)(bb * NHEAD + hh);
                if (s == 0)      Qb[(headidx * LSEQ + l) * HDIM + d] = bv;
                else if (s == 1) Kb[(headidx * LSEQ + l) * HDIM + d] = bv;
                else             Vt[(headidx * HDIM + d) * LSEQ + l] = bv;
            }
        }
    }
}

// ---------------------------------------------------------------------------
// Kernel 2: banded local attention with online softmax (unchanged math);
// output written as bf16 rows so the projection GEMM consumes it directly.
// ---------------------------------------------------------------------------
__global__ __launch_bounds__(32) void local_attn_kernel(
    const unsigned short* __restrict__ Qb,
    const unsigned short* __restrict__ Kb,
    const unsigned short* __restrict__ Vt,
    unsigned short* __restrict__ attn_out)    // [B*L, 512] bf16
{
    __shared__ unsigned short ldsP[16 * 32];

    const int lane = threadIdx.x;
    const int m = lane & 15, h = lane >> 4;
    const int qt   = blockIdx.x & 255;
    const int bh   = blockIdx.x >> 8;
    const int head = bh & (NHEAD - 1);
    const int bb   = bh >> 3;
    const int qstart = qt * 16;

    const unsigned short* Qh = Qb + (size_t)bh * LSEQ * HDIM;
    const unsigned short* Kh = Kb + (size_t)bh * LSEQ * HDIM;
    const unsigned short* Vh = Vt + (size_t)bh * HDIM * LSEQ;

    const unsigned short* qrow = Qh + (size_t)(qstart + m) * HDIM;
    v16bf aq0 = load_a_bf16(qrow, h);
    v16bf aq1 = load_a_bf16(qrow + 32, h);

    v8f o[4] = {{0,0,0,0,0,0,0,0}, {0,0,0,0,0,0,0,0},
                {0,0,0,0,0,0,0,0}, {0,0,0,0,0,0,0,0}};
    float rmax[8], rsum[8];
#pragma unroll
    for (int v = 0; v < 8; ++v) { rmax[v] = -__builtin_inff(); rsum[v] = 0.0f; }

    for (int kt2 = 0; kt2 < 5; ++kt2) {
        const int ktbase = qstart - HALFW + kt2 * 32;

        // ---- S = Q K^T for two 16-key sub-tiles ----
        float s[2][8];
#pragma unroll
        for (int c = 0; c < 2; ++c) {
            int keyslot = ktbase + c * 16 + m;
            int krow = min(max(keyslot, 0), LSEQ - 1);
            const unsigned short* kr = Kh + (size_t)krow * HDIM;
            v16bf bk0 = load_b_bf16(kr, h);
            v16bf bk1 = load_b_bf16(kr + 32, h);
            v8f sc = {0,0,0,0,0,0,0,0};
            sc = wmma_bf16(aq0, bk0, sc);
            sc = wmma_bf16(aq1, bk1, sc);
#pragma unroll
            for (int v = 0; v < 8; ++v) {
                int key = ktbase + c * 16 + m;
                int q   = qstart + v + 8 * h;
                int dlt = q - key;
                bool valid = (key >= 0) && (key < LSEQ) &&
                             (dlt <= HALFW) && (dlt >= -HALFW);
                s[c][v] = valid ? sc[v] * ATTN_SCALE : -__builtin_inff();
            }
        }

        // ---- online softmax update (row = v + 8*h, reduce over lane halves) ----
#pragma unroll
        for (int v = 0; v < 8; ++v) {
            float tm = fmaxf(s[0][v], s[1][v]);
            tm = fmaxf(tm, __shfl_xor(tm, 1));
            tm = fmaxf(tm, __shfl_xor(tm, 2));
            tm = fmaxf(tm, __shfl_xor(tm, 4));
            tm = fmaxf(tm, __shfl_xor(tm, 8));
            float newmax = fmaxf(rmax[v], tm);
            float mg  = (newmax == -__builtin_inff()) ? 0.0f : newmax;
            float fac = __expf(rmax[v] - mg);
            float p0 = __expf(s[0][v] - mg);
            float p1 = __expf(s[1][v] - mg);
            float ps = p0 + p1;
            ps += __shfl_xor(ps, 1);
            ps += __shfl_xor(ps, 2);
            ps += __shfl_xor(ps, 4);
            ps += __shfl_xor(ps, 8);
            rsum[v] = rsum[v] * fac + ps;
            rmax[v] = newmax;
#pragma unroll
            for (int dt = 0; dt < 4; ++dt) o[dt][v] *= fac;
            ldsP[(v + 8 * h) * 32 + m]      = f32_to_bf16(p0);
            ldsP[(v + 8 * h) * 32 + 16 + m] = f32_to_bf16(p1);
        }
        __syncthreads();
        v16bf ap = load_a_bf16(&ldsP[m * 32], h);   // P as 16x32 A operand
        __syncthreads();

        // ---- O += P * V  (B operand from V^T rows: contiguous keys) ----
#pragma unroll
        for (int dt = 0; dt < 4; ++dt) {
            int d  = dt * 16 + m;
            int k0 = min(max(ktbase + h * 8, 0), LSEQ - 8);
            int k1 = min(max(ktbase + 16 + h * 8, 0), LSEQ - 8);
            const unsigned short* vr = Vh + (size_t)d * LSEQ;
            BF16Tile t;
            t.q[0] = *reinterpret_cast<const v4u*>(vr + k0);
            t.q[1] = *reinterpret_cast<const v4u*>(vr + k1);
            o[dt] = wmma_bf16(ap, t.v, o[dt]);
        }
    }

    // ---- normalize and write bf16 [B*L, C], column = head*64 + d ----
#pragma unroll
    for (int v = 0; v < 8; ++v) {
        float inv = (rsum[v] > 0.0f) ? (1.0f / rsum[v]) : 0.0f;
        int row = bb * LSEQ + qstart + v + 8 * h;
        unsigned short* orow = attn_out + (size_t)row * CDIM + head * HDIM;
#pragma unroll
        for (int dt = 0; dt < 4; ++dt)
            orow[dt * 16 + m] = f32_to_bf16(o[dt][v] * inv);
    }
}

// ---------------------------------------------------------------------------
// Kernel 3: output projection GEMM (bf16 in, f32 out).  32x64 per wave.
// ---------------------------------------------------------------------------
__global__ __launch_bounds__(32) void proj_gemm_kernel(
    const unsigned short* __restrict__ abf,   // [B*L, 512] bf16
    const unsigned short* __restrict__ wT,    // [512, 512] bf16
    float* __restrict__ out)                  // [B*L, 512] f32
{
    const int lane = threadIdx.x;
    const int m = lane & 15, h = lane >> 4;
    const int rt    = blockIdx.x / 8;         // 8 = 512/64 col strips
    const int nt    = blockIdx.x % 8;
    const int rbase = rt * 32;
    const int nbase = nt * 64;

    GemmAcc g;
    gemm_32x64(abf + (size_t)(rbase + m) * CDIM,
               abf + (size_t)(rbase + 16 + m) * CDIM,
               wT, nbase, m, h, g);

#pragma unroll
    for (int t = 0; t < 4; ++t) {
#pragma unroll
        for (int v = 0; v < 8; ++v) {
            int col = nbase + t * 16 + m;
            out[(size_t)(rbase + v + 8 * h) * CDIM + col]      = g.a0[t][v];
            out[(size_t)(rbase + 16 + v + 8 * h) * CDIM + col] = g.a1[t][v];
        }
    }
}

// ---------------------------------------------------------------------------
extern "C" void kernel_launch(void* const* d_in, const int* in_sizes, int n_in,
                              void* d_out, int out_size, void* d_ws, size_t ws_size,
                              hipStream_t stream) {
    (void)in_sizes; (void)n_in; (void)out_size; (void)ws_size;
    const float* x      = (const float*)d_in[0];   // [B,L,C]
    const float* w_qkv  = (const float*)d_in[1];   // [C, 3C]
    const float* w_proj = (const float*)d_in[2];   // [C, C]

    char* ws = (char*)d_ws;
    unsigned short* wqkvT  = (unsigned short*)(ws);                 // 1536*512 bf16
    unsigned short* wprojT = (unsigned short*)(ws + 1572864);       //  512*512 bf16
    unsigned short* xbf    = (unsigned short*)(ws + 2097152);       // 8192*512 bf16
    unsigned short* Qb     = (unsigned short*)(ws + 10485760);      // 2*8*4096*64 bf16
    unsigned short* Kb     = (unsigned short*)(ws + 18874368);
    unsigned short* Vt     = (unsigned short*)(ws + 27262976);
    unsigned short* attnbf = (unsigned short*)(ws + 35651584);      // 8192*512 bf16
    float*          out    = (float*)d_out;                         // [B*L, 512]

    // 786432 + 262144 + 4194304 = 5242880 elements -> 20480 blocks of 256
    prep_inputs_kernel<<<20480, 256, 0, stream>>>(w_qkv, w_proj, x,
                                                  wqkvT, wprojT, xbf);
    qkv_gemm_kernel<<<256 * 24, 32, 0, stream>>>(xbf, wqkvT, Qb, Kb, Vt);
    local_attn_kernel<<<2 * 8 * 256, 32, 0, stream>>>(Qb, Kb, Vt, attnbf);
    proj_gemm_kernel<<<256 * 8, 32, 0, stream>>>(attnbf, wprojT, out);
}
